// LSTM_455266534066
// MI455X (gfx1250) — matrix operand order repaired and checked by reading, not compiled, not run on hardware
//
#include <hip/hip_runtime.h>
#include <hip/hip_bf16.h>
#include <stdint.h>

// Problem dims (fixed by the reference)
#define TT 512
#define BB 64
#define II 1024
#define HH 1024
#define GG (4 * HH)      // 4096 gate columns
#define MM (TT * BB)     // 32768 rows of the big GEMM

// LDS A-tile: 64 rows x 256 K-halves, padded by 8 halves (TDM pad: 4 DWORDs
// every 128 DWORDs) so the row stride (528 B) stays 16B-aligned for
// ds_load_b128 while spreading the 16 lanes' row accesses 4 banks apart
// (528 mod 256 = 16B) -> conflict free across 64 banks.
#define KP   256
#define KPAD 264
#define AS_BYTES (BB * KPAD * 2)      // 33792
#define GXS_BYTES (4 * BB * 16 * 4)   // 16384

typedef __attribute__((ext_vector_type(16))) __bf16 v16bf;
typedef __attribute__((ext_vector_type(8)))  __bf16 v8bf;
typedef __attribute__((ext_vector_type(8)))  float  v8f;
typedef __attribute__((ext_vector_type(4)))  unsigned int u32x4;
typedef __attribute__((ext_vector_type(8)))  int i32x8;
typedef __attribute__((ext_vector_type(4)))  int i32x4;

// ---------------------------------------------------------------------------
// fp32 -> bf16 conversion
// ---------------------------------------------------------------------------
__global__ void lstm_f32_to_bf16(const float* __restrict__ in,
                                 __bf16* __restrict__ out, size_t n) {
    size_t i = (size_t)blockIdx.x * blockDim.x + threadIdx.x;
    if (i < n) out[i] = (__bf16)in[i];
}

__global__ void lstm_bias_sum(const float* __restrict__ bx,
                              const float* __restrict__ bh,
                              float* __restrict__ bs, int n) {
    int i = blockIdx.x * blockDim.x + threadIdx.x;
    if (i < n) bs[i] = bx[i] + bh[i];
}

// ---------------------------------------------------------------------------
// Fragment builders for V_WMMA_F32_16X16X32_BF16 (wave32 layouts, ISA 7.12.2)
// ---------------------------------------------------------------------------
__device__ __forceinline__ v16bf make_frag(v8bf lo, v8bf hi) {
    v16bf f;
#pragma unroll
    for (int i = 0; i < 8; ++i) { f[i] = lo[i]; f[i + 8] = hi[i]; }
    return f;
}

__device__ __forceinline__ v8f wmma_acc(v16bf A, v16bf B, v8f acc) {
    return __builtin_amdgcn_wmma_f32_16x16x32_bf16(
        false, A, false, B, (short)0, acc, false, false);
}

// ---------------------------------------------------------------------------
// Phase 1: gx[M,G] = Xb[M,I] @ Wxb[G,I]^T + bsum[G]
// 8 waves / block; wave w computes a 16x64 strip (1 A fragment reused across
// 4 B tiles -> 4 accumulators). Block tile 128x64; grid (G/64, M/128).
// x(bf16, 64MB) + W_x(bf16, 8MB) are L2-resident on MI455X (192MB L2).
// ---------------------------------------------------------------------------
__global__ __launch_bounds__(256)
void lstm_gemm_gx(const __bf16* __restrict__ Xb,
                  const __bf16* __restrict__ Wxb,
                  const float* __restrict__ bsum,
                  float* __restrict__ gx) {
    const int lane = threadIdx.x & 31;
    const int wave = threadIdx.x >> 5;            // 0..7
    const int row0 = blockIdx.y * 128 + wave * 16;
    const int col0 = blockIdx.x * 64;

    const int arow = row0 + (lane & 15);
    const int asel = (lane < 16) ? 0 : 8;
    const int bsel = (lane < 16) ? 0 : 8;

    const __bf16* aptr = Xb + (size_t)arow * II;
    const __bf16* bptr[4];
#pragma unroll
    for (int c = 0; c < 4; ++c)
        bptr[c] = Wxb + (size_t)(col0 + c * 16 + (lane & 15)) * II;

    v8f acc[4] = {};
#pragma unroll 2
    for (int k = 0; k < II; k += 32) {
        v8bf a0 = *(const v8bf*)(aptr + k + asel);
        v8bf a1 = *(const v8bf*)(aptr + k + 16 + asel);
        v16bf A = make_frag(a0, a1);
#pragma unroll
        for (int c = 0; c < 4; ++c) {
            v8bf b0 = *(const v8bf*)(bptr[c] + k + bsel);
            v8bf b1 = *(const v8bf*)(bptr[c] + k + bsel + 16);
            acc[c] = wmma_acc(A, make_frag(b0, b1), acc[c]);
        }
    }

    const int rofs = (lane < 16) ? 0 : 8;
#pragma unroll
    for (int c = 0; c < 4; ++c) {
        const int bcol = col0 + c * 16 + (lane & 15);
        const float bias = bsum[bcol];
#pragma unroll
        for (int i = 0; i < 8; ++i)
            gx[(size_t)(row0 + rofs + i) * GG + bcol] = acc[c][i] + bias;
    }
}

// ---------------------------------------------------------------------------
// Phase 2: one LSTM step.
// 64 blocks x 512 threads (16 waves); wave = (gate g, row block r).
//  * gx gate-bias tile (64x64 f32, 16KB) is async-loaded to LDS at kernel
//    entry (global_load_async_to_lds_b128) and waited on only AFTER the
//    recurrent GEMM -> hides the gx HBM stream behind WMMA work.
//  * h (64x1024 bf16) is staged in four 64x256 tiles by the Tensor Data
//    Mover (tensor_load_to_lds, one DMA per workgroup, LDS row padding via
//    D# pad fields), completion via s_wait_tensorcnt + barrier.
//  * WMMA A fragments from LDS (ds_load_b128), B from L2-resident W_h.
//  * Gate tiles exchanged via LDS (aliasing the dead A tile), then a fused
//    sigmoid/tanh cell update.
// ---------------------------------------------------------------------------
__global__ __launch_bounds__(512)
void lstm_step(const float* __restrict__ gx_t,    // [B, 4H] slice for step t
               const __bf16* __restrict__ hin,    // [B, H] bf16 (prev h)
               const __bf16* __restrict__ Whb,    // [4H, H] bf16
               float* __restrict__ cws,           // [B, H] fp32 cell state
               float* __restrict__ hout_f32,      // d_out slice [B, H]
               __bf16* __restrict__ hout_bf) {    // [B, H] bf16 (next h)
    __shared__ __align__(16) char smem[AS_BYTES + GXS_BYTES];   // 49 KB
    __bf16 (*As)[KPAD] = (__bf16(*)[KPAD])smem;                 // 33 KB
    float* gxs = (float*)(smem + AS_BYTES);                     // 16 KB
    float (*gtile)[4][16][16] = (float(*)[4][16][16])smem;      // aliases As

    const int tid  = threadIdx.x;
    const int lane = tid & 31;
    const int wave = tid >> 5;                    // 0..15
    const int g    = wave >> 2;                   // gate 0..3
    const int r    = wave & 3;                    // row block 0..3
    const int col0 = blockIdx.x * 16;             // column within H

    const uint32_t lds_as  = (uint32_t)(size_t)&As[0][0];
    const uint32_t lds_gxs = (uint32_t)(size_t)gxs;

    // ---- prefetch gx tile to LDS, waited on after the GEMM ----
    // gxs layout: [gate][row][16 cols] f32. 1024 x 16B transfers, 2/thread.
    {
        const uint64_t gxbase = (uint64_t)(uintptr_t)gx_t;
#pragma unroll
        for (int p = 0; p < 2; ++p) {
            const int xfer  = tid + p * 512;
            const int chunk = xfer >> 2;          // (g*64 + row), 0..255
            const int q     = xfer & 3;
            const int cg    = chunk >> 6;
            const int crow  = chunk & 63;
            const uint32_t moff = (uint32_t)((crow * GG + cg * HH + col0) * 4 + q * 16);
            const uint32_t loff = lds_gxs + (uint32_t)(chunk * 64 + q * 16);
            asm volatile("global_load_async_to_lds_b128 %0, %1, %2"
                         :: "v"(loff), "v"(moff), "s"(gxbase)
                         : "memory");
        }
    }

    const int arow = r * 16 + (lane & 15);
    const int asel = (lane < 16) ? 0 : 8;
    const int bsel = (lane < 16) ? 0 : 8;
    const int gcol = g * HH + col0 + (lane & 15); // column within 4H
    const __bf16* bptr = Whb + (size_t)gcol * HH;
    const uint64_t hbase = (uint64_t)(uintptr_t)hin;

    v8f acc = {};
    for (int kk = 0; kk < HH; kk += KP) {
        // ---- TDM: DMA 64x256 bf16 tile of h into padded LDS ----
        if (wave == 0) {
            const uint64_t ga = hbase + (uint64_t)kk * 2u;
            u32x4 g0;
            g0[0] = 1u;                                    // count=1 (valid D#)
            g0[1] = lds_as;                                // lds_addr
            g0[2] = (unsigned)ga;                          // global_addr[31:0]
            g0[3] = (unsigned)((ga >> 32) & 0x1FFFFFFu)    // global_addr[56:32]
                  | (2u << 30);                            // type=2 (image)
            i32x8 g1;
            g1[0] = (int)((1u << 16)      // data_size=2B
                        | (1u << 20)      // pad_enable
                        | (6u << 22)      // pad_interval: every 128 DWORDs
                        | (3u << 25));    // pad_amount: 4 DWORDs
            g1[1] = (int)(1024u << 16);   // tensor_dim0 = 1024 (low16)
            g1[2] = (int)(64u << 16);     // dim0 hi=0 | tensor_dim1=64 (low16)
            g1[3] = (int)(256u << 16);    // dim1 hi=0 | tile_dim0=256
            g1[4] = 64;                   // tile_dim1=64 | tile_dim2=0
            g1[5] = 1024;                 // tensor_dim0_stride (low 32)
            g1[6] = 0;                    // stride hi | tensor_dim1_stride lo
            g1[7] = 0;
            i32x4 gz4 = {0, 0, 0, 0};
            i32x8 gz8 = {0, 0, 0, 0, 0, 0, 0, 0};
            __builtin_amdgcn_tensor_load_to_lds(g0, g1, gz4, gz4, gz8, 0);
            __builtin_amdgcn_s_wait_tensorcnt(0);
        }
        __syncthreads();

        // ---- 8 WMMA K-steps out of LDS ----
#pragma unroll 2
        for (int k = 0; k < KP; k += 32) {
            v8bf a0 = *(const v8bf*)&As[arow][k + asel];
            v8bf a1 = *(const v8bf*)&As[arow][k + 16 + asel];
            v8bf b0 = *(const v8bf*)(bptr + kk + k + bsel);
            v8bf b1 = *(const v8bf*)(bptr + kk + k + bsel + 16);
            acc = wmma_acc(make_frag(a0, a1), make_frag(b0, b1), acc);
        }
        __syncthreads();   // protect As before next pass overwrites it
    }

    // ---- gx tile must be in LDS by now (it almost surely is) ----
    asm volatile("s_wait_asynccnt 0" ::: "memory");
    __syncthreads();

    // ---- exchange gate tiles through LDS (aliases the dead As region) ----
    const int rofs = (lane < 16) ? 0 : 8;
#pragma unroll
    for (int i = 0; i < 8; ++i) {
        const int row = r * 16 + rofs + i;
        gtile[g][r][rofs + i][lane & 15] =
            acc[i] + gxs[(g * BB + row) * 16 + (lane & 15)];
    }
    __syncthreads();

    // ---- fused cell update: 64 rows x 16 cols, 2 cells per thread ----
    for (int cell = tid; cell < BB * 16; cell += 512) {
        const int row = cell >> 4;
        const int n   = cell & 15;
        const int rb2 = row >> 4, m = row & 15;
        const float xi = gtile[0][rb2][m][n];
        const float xf = gtile[1][rb2][m][n];
        const float xo = gtile[2][rb2][m][n];
        const float xn = gtile[3][rb2][m][n];
        const float ig = 1.0f / (1.0f + __expf(-xi));
        const float fg = 1.0f / (1.0f + __expf(-xf));
        const float og = 1.0f / (1.0f + __expf(-xo));
        const float ng = tanhf(xn);
        const size_t idx = (size_t)row * HH + col0 + n;
        const float c = fg * cws[idx] + ig * ng;
        const float h = og * tanhf(c);
        cws[idx]      = c;
        hout_f32[idx] = h;
        hout_bf[idx]  = (__bf16)h;
    }
}

// ---------------------------------------------------------------------------
// Host-side orchestration
// ---------------------------------------------------------------------------
extern "C" void kernel_launch(void* const* d_in, const int* in_sizes, int n_in,
                              void* d_out, int out_size, void* d_ws, size_t ws_size,
                              hipStream_t stream) {
    const float* x   = (const float*)d_in[0];   // [T, B, I]
    const float* h0  = (const float*)d_in[1];   // [B, H]
    const float* c0  = (const float*)d_in[2];   // [B, H]
    const float* W_x = (const float*)d_in[3];   // [4H, I]
    const float* W_h = (const float*)d_in[4];   // [4H, H]
    const float* b_x = (const float*)d_in[5];   // [4H]
    const float* b_h = (const float*)d_in[6];   // [4H]
    float* out = (float*)d_out;                 // outputs[T,B,H] ++ h[B,H] ++ c[B,H]

    // Workspace layout
    char* ws = (char*)d_ws;
    size_t off = 0;
    float*  gx   = (float*)(ws + off);  off += (size_t)MM * GG * sizeof(float);   // 512 MB
    __bf16* Xb   = (__bf16*)(ws + off); off += (size_t)MM * II * sizeof(__bf16);  // 64 MB
    __bf16* Wxb  = (__bf16*)(ws + off); off += (size_t)GG * II * sizeof(__bf16);  // 8 MB
    __bf16* Whb  = (__bf16*)(ws + off); off += (size_t)GG * HH * sizeof(__bf16);  // 8 MB
    float*  bsum = (float*)(ws + off);  off += (size_t)GG * sizeof(float);
    __bf16* hbf0 = (__bf16*)(ws + off); off += (size_t)BB * HH * sizeof(__bf16);
    __bf16* hbf1 = (__bf16*)(ws + off); off += (size_t)BB * HH * sizeof(__bf16);
    float*  cws  = (float*)(ws + off);  off += (size_t)BB * HH * sizeof(float);
    (void)ws_size; (void)in_sizes; (void)n_in; (void)out_size;

    // --- conversions / precompute ---
    {
        const size_t nx = (size_t)MM * II;
        lstm_f32_to_bf16<<<(unsigned)((nx + 255) / 256), 256, 0, stream>>>(x, Xb, nx);
        const size_t nwx = (size_t)GG * II;
        lstm_f32_to_bf16<<<(unsigned)((nwx + 255) / 256), 256, 0, stream>>>(W_x, Wxb, nwx);
        const size_t nwh = (size_t)GG * HH;
        lstm_f32_to_bf16<<<(unsigned)((nwh + 255) / 256), 256, 0, stream>>>(W_h, Whb, nwh);
        const size_t nh = (size_t)BB * HH;
        lstm_f32_to_bf16<<<(unsigned)((nh + 255) / 256), 256, 0, stream>>>(h0, hbf0, nh);
        lstm_bias_sum<<<GG / 256, 256, 0, stream>>>(b_x, b_h, bsum, GG);
        (void)hipMemcpyAsync(cws, c0, (size_t)BB * HH * sizeof(float),
                             hipMemcpyDeviceToDevice, stream);
    }

    // --- Phase 1: big input-projection GEMM (bf16 WMMA, 16x64 per wave) ---
    lstm_gemm_gx<<<dim3(GG / 64, MM / 128), 256, 0, stream>>>(Xb, Wxb, bsum, gx);

    // --- Phase 2: sequential recurrence, one kernel per timestep ---
    for (int t = 0; t < TT; ++t) {
        const __bf16* hin = (t & 1) ? hbf1 : hbf0;
        __bf16* hob       = (t & 1) ? hbf0 : hbf1;
        lstm_step<<<HH / 16, 512, 0, stream>>>(
            gx + (size_t)t * BB * GG, hin, Whb, cws,
            out + (size_t)t * BB * HH, hob);
    }

    // --- tail: final h (== outputs[T-1]) and c into d_out ---
    (void)hipMemcpyAsync(out + (size_t)TT * BB * HH,
                         out + (size_t)(TT - 1) * BB * HH,
                         (size_t)BB * HH * sizeof(float),
                         hipMemcpyDeviceToDevice, stream);
    (void)hipMemcpyAsync(out + (size_t)TT * BB * HH + (size_t)BB * HH,
                         cws,
                         (size_t)BB * HH * sizeof(float),
                         hipMemcpyDeviceToDevice, stream);
}
